// module_18090402251122
// MI455X (gfx1250) — compile-verified, matrix operand order checked by reading, not run on hardware
//
#include <hip/hip_runtime.h>
#include <hip/hip_bf16.h>

// MLP forward: x(2M,16) -> [Linear(16,50)+ReLU] -> 14x[Linear(50,50)+ReLU] -> Linear(50,4)
// Strategy: transposed-activation f16 WMMA (D = W * h^T), f32 accumulate.
// D-layout of h^T is lane-for-lane identical to the B-fragment layout of the
// next layer's WMMA, so inter-layer handoff is pure per-lane relu+cvt.
// Bias folded via augmented constant-1 input column (K=16 for layer0, K=50 else).

typedef _Float16 v16h __attribute__((ext_vector_type(16)));
typedef float    v8f  __attribute__((ext_vector_type(8)));
typedef float    v4f  __attribute__((ext_vector_type(4)));

#define NBT 4            // batch tiles (16 rows each) per wave -> 64 rows/wave
#define N_HIDDEN 14
#define NFRAG (4 + N_HIDDEN*8 + 2)   // 118 weight fragments, 1KB each

// ---------------------------------------------------------------------------
// Prep: swizzle weights+bias into per-lane WMMA A-fragment layout (f16).
// A fragment (16x32, M=out-neuron tile, K=in-neuron chunk), per lane 16 f16:
//   m = mt*16 + (lane&15), khalf = lane>>4
//   elem j<8 : k = kb + 8*khalf + j
//   elem j>=8: k = kb + 16 + 8*khalf + (j-8)
// Bias lives at k == in_dim (augmented column). Everything else padded 0.
// ---------------------------------------------------------------------------
__global__ void prep_frags(const float* __restrict__ W0, const float* __restrict__ b0,
                           const float* __restrict__ Wh, const float* __restrict__ bh,
                           const float* __restrict__ Wo, const float* __restrict__ bo,
                           _Float16* __restrict__ frags) {
    int idx = blockIdx.x * blockDim.x + threadIdx.x;     // one thread per f16 element
    if (idx >= NFRAG * 32 * 16) return;
    int elem = idx & 15;
    int lane = (idx >> 4) & 31;
    int frag = idx >> 9;

    int mt, c, kind, h = 0;
    if (frag < 4)            { kind = 0; mt = frag; c = 0; }
    else if (frag < 4 + N_HIDDEN*8) {
        int f = frag - 4; h = f >> 3; int r = f & 7;
        kind = 1; mt = r >> 1; c = r & 1;
    } else                   { kind = 2; mt = 0; c = frag - (4 + N_HIDDEN*8); }

    int khalf = lane >> 4;
    int m  = mt*16 + (lane & 15);
    int kb = c*32;
    int k  = (elem < 8) ? (kb + 8*khalf + elem) : (kb + 16 + 8*khalf + (elem - 8));

    float val = 0.0f;
    if (kind == 0) {              // layer 0: W0(50,16), bias at k==16
        if (m < 50) { if (k < 16) val = W0[m*16 + k]; else if (k == 16) val = b0[m]; }
    } else if (kind == 1) {       // hidden: Wh(h,50,50), bias at k==50
        if (m < 50) { if (k < 50) val = Wh[(h*50 + m)*50 + k]; else if (k == 50) val = bh[h*50 + m]; }
    } else {                      // output: Wo(4,50), bias at k==50
        if (m < 4)  { if (k < 50) val = Wo[m*50 + k];          else if (k == 50) val = bo[m]; }
    }
    frags[frag*512 + lane*16 + elem] = (_Float16)val;
}

// ---------------------------------------------------------------------------
// Main kernel: one wave processes 64 batch rows through the whole network.
// acc[bt][mt] : f32 D tiles of h^T (neurons mt*16.. , batch cols 0..15)
// bfr[bt][c]  : f16 B fragments for current layer's K chunk c
// ---------------------------------------------------------------------------
__global__ __launch_bounds__(256) void mlp_fwd(const float* __restrict__ x,
                                               const _Float16* __restrict__ frags,
                                               float* __restrict__ out, int batch) {
    const int lane   = threadIdx.x & 31;
    const int waveId = blockIdx.x * (blockDim.x >> 5) + (threadIdx.x >> 5);
    const long long tileBase = (long long)waveId * (16 * NBT);
    if (tileBase + 16 * NBT > batch) return;   // wave-uniform guard (EXEC stays full)

    const int mlocal = lane & 15;
    const int khalf  = lane >> 4;

    v8f  acc[NBT][4];
    v16h bfr[NBT][2];
    const v8f z8 = 0.0f;

    // ---------------- Layer 0: B from x (K=16 real + bias slot K=16..pad 32) ----
#pragma unroll
    for (int bt = 0; bt < NBT; ++bt) {
        const float* xr = x + (size_t)(tileBase + bt*16 + mlocal) * 16 + khalf*8;
        v4f x0 = *(const v4f*)xr;
        v4f x1 = *(const v4f*)(xr + 4);
        v16h b;
        b[0] = (_Float16)x0.x; b[1] = (_Float16)x0.y; b[2] = (_Float16)x0.z; b[3] = (_Float16)x0.w;
        b[4] = (_Float16)x1.x; b[5] = (_Float16)x1.y; b[6] = (_Float16)x1.z; b[7] = (_Float16)x1.w;
        b[8] = (khalf == 0) ? (_Float16)1.0f : (_Float16)0.0f;   // bias slot K=16
#pragma unroll
        for (int j = 9; j < 16; ++j) b[j] = (_Float16)0.0f;
        bfr[bt][0] = b;
    }
#pragma unroll
    for (int mt = 0; mt < 4; ++mt) {
        v16h wf = *(const v16h*)(frags + mt*512 + lane*16);
#pragma unroll
        for (int bt = 0; bt < NBT; ++bt)
            acc[bt][mt] = __builtin_amdgcn_wmma_f32_16x16x32_f16(
                false, wf, false, bfr[bt][0], (short)0, z8, false, false);
    }

    // ---------------- 14 hidden layers --------------------------------------
    for (int h = 0; h < N_HIDDEN; ++h) {
        // Build next-layer B fragments: pure per-lane relu + cvt from D tiles.
#pragma unroll
        for (int bt = 0; bt < NBT; ++bt) {
#pragma unroll
            for (int c = 0; c < 2; ++c) {
                v16h b;
#pragma unroll
                for (int j = 0; j < 8; ++j) b[j]     = (_Float16)fmaxf(acc[bt][2*c  ][j], 0.0f);
#pragma unroll
                for (int j = 0; j < 8; ++j) b[8 + j] = (_Float16)fmaxf(acc[bt][2*c+1][j], 0.0f);
                bfr[bt][c] = b;
            }
            // bias slot K=50 -> chunk1 elem10, lanes 0..15 only
            bfr[bt][1][10] = (khalf == 0) ? (_Float16)1.0f : bfr[bt][1][10];
        }
        const _Float16* lf = frags + (size_t)(4 + h*8) * 512;
#pragma unroll
        for (int mt = 0; mt < 4; ++mt) {
            v16h wf0 = *(const v16h*)(lf + (mt*2 + 0)*512 + lane*16);
            v16h wf1 = *(const v16h*)(lf + (mt*2 + 1)*512 + lane*16);
#pragma unroll
            for (int bt = 0; bt < NBT; ++bt)
                acc[bt][mt] = __builtin_amdgcn_wmma_f32_16x16x32_f16(
                    false, wf0, false, bfr[bt][0], (short)0, z8, false, false);
#pragma unroll
            for (int bt = 0; bt < NBT; ++bt)
                acc[bt][mt] = __builtin_amdgcn_wmma_f32_16x16x32_f16(
                    false, wf1, false, bfr[bt][1], (short)0, acc[bt][mt], false, false);
        }
    }

    // ---------------- Output layer (no relu), 1 M-tile, 2 K chunks ----------
#pragma unroll
    for (int bt = 0; bt < NBT; ++bt) {
#pragma unroll
        for (int c = 0; c < 2; ++c) {
            v16h b;
#pragma unroll
            for (int j = 0; j < 8; ++j) b[j]     = (_Float16)fmaxf(acc[bt][2*c  ][j], 0.0f);
#pragma unroll
            for (int j = 0; j < 8; ++j) b[8 + j] = (_Float16)fmaxf(acc[bt][2*c+1][j], 0.0f);
            bfr[bt][c] = b;
        }
        bfr[bt][1][10] = (khalf == 0) ? (_Float16)1.0f : bfr[bt][1][10];
    }
    {
        const _Float16* lf = frags + (size_t)(4 + N_HIDDEN*8) * 512;
        v16h wf0 = *(const v16h*)(lf + 0*512 + lane*16);
        v16h wf1 = *(const v16h*)(lf + 1*512 + lane*16);
        v8f o[NBT];
#pragma unroll
        for (int bt = 0; bt < NBT; ++bt)
            o[bt] = __builtin_amdgcn_wmma_f32_16x16x32_f16(
                false, wf0, false, bfr[bt][0], (short)0, z8, false, false);
#pragma unroll
        for (int bt = 0; bt < NBT; ++bt)
            o[bt] = __builtin_amdgcn_wmma_f32_16x16x32_f16(
                false, wf1, false, bfr[bt][1], (short)0, o[bt], false, false);

        // D: lanes 0..15 hold neurons 0..7 in v[0..7] for batch col = lane.
        if (lane < 16) {
#pragma unroll
            for (int bt = 0; bt < NBT; ++bt) {
                v4f r; r.x = o[bt][0]; r.y = o[bt][1]; r.z = o[bt][2]; r.w = o[bt][3];
                *(v4f*)(out + (size_t)(tileBase + bt*16 + lane) * 4) = r;
            }
        }
    }
}

// ---------------------------------------------------------------------------
extern "C" void kernel_launch(void* const* d_in, const int* in_sizes, int n_in,
                              void* d_out, int out_size, void* d_ws, size_t ws_size,
                              hipStream_t stream) {
    const float* x  = (const float*)d_in[0];
    const float* W0 = (const float*)d_in[1];
    const float* b0 = (const float*)d_in[2];
    const float* Wh = (const float*)d_in[3];
    const float* bh = (const float*)d_in[4];
    const float* Wo = (const float*)d_in[5];
    const float* bo = (const float*)d_in[6];
    float* out = (float*)d_out;
    const int batch = in_sizes[0] / 16;

    _Float16* frags = (_Float16*)d_ws;   // needs NFRAG*1KB ~= 118KB of d_ws

    // 1) swizzle weights/bias into WMMA A-fragment layout (runs every launch;
    //    deterministic, graph-capture safe, ordered on `stream`)
    const int prepElems = NFRAG * 32 * 16;
    prep_frags<<<(prepElems + 255) / 256, 256, 0, stream>>>(W0, b0, Wh, bh, Wo, bo, frags);

    // 2) main fused MLP: 64 rows per wave, 8 waves per block
    const long long waves  = (batch + 16*NBT - 1) / (16*NBT);
    const int blocks = (int)((waves + 7) / 8);
    mlp_fwd<<<blocks, 256, 0, stream>>>(x, frags, out, batch);
}
